// VAE_gamma_86139864089344
// MI455X (gfx1250) — matrix operand order repaired and checked by reading, not compiled, not run on hardware
//
#include <hip/hip_runtime.h>
#include <hip/hip_bf16.h>
#include <stdint.h>

// ---------------------------------------------------------------------------
// Types for WMMA fragments (wave32, v_wmma_f32_16x16x32_f16)
// ---------------------------------------------------------------------------
typedef __attribute__((ext_vector_type(16))) _Float16 v16h;
typedef __attribute__((ext_vector_type(8)))  _Float16 v8h;
typedef __attribute__((ext_vector_type(4)))  _Float16 v4h;
typedef __attribute__((ext_vector_type(8)))  float    v8f;
typedef __attribute__((ext_vector_type(4)))  float    v4f;
typedef __attribute__((ext_vector_type(4)))  int      v4i;

#define EPSF 1e-6f

#define AS_GLOBAL __attribute__((address_space(1)))
#define AS_LOCAL  __attribute__((address_space(3)))

// CDNA5 async global->LDS copy path (ASYNCcnt-tracked, no VGPR round trip).
// Probe-verified: param0 = int4 addrspace(1)*, param1 = int4 addrspace(3)*,
// then (imm offset, imm cpol). Emits global_load_async_to_lds_b128.
#if __has_builtin(__builtin_amdgcn_global_load_async_to_lds_b128) && \
    __has_builtin(__builtin_amdgcn_s_wait_asynccnt)
#define USE_ASYNC_LDS 1
typedef AS_GLOBAL v4i* gptr_b128;
typedef AS_LOCAL  v4i* lptr_b128;
#else
#define USE_ASYNC_LDS 0
#endif

// ---------------------------------------------------------------------------
// Tiled transpose + fp32 -> fp16 convert: src[K][N] -> dst[N][K].
// Pre-transposed weights make GEMM B-tiles K-contiguous -> async b128 copies.
// ---------------------------------------------------------------------------
__global__ __launch_bounds__(256) void transpose_f32_to_f16(const float* __restrict__ src,
                                                            _Float16* __restrict__ dst,
                                                            int K, int N) {
  __shared__ float tile[32][33];
  const int kt = blockIdx.x * 32, nt = blockIdx.y * 32;
  const int tx = threadIdx.x & 31, ty = threadIdx.x >> 5;  // 32 x 8
#pragma unroll
  for (int r = 0; r < 32; r += 8)
    tile[ty + r][tx] = src[(size_t)(kt + ty + r) * N + nt + tx];
  __syncthreads();
#pragma unroll
  for (int r = 0; r < 32; r += 8)
    dst[(size_t)(nt + ty + r) * K + kt + tx] = (_Float16)tile[tx][ty + r];
}

// Pack fc41_w (K x 64) and fc42_w (K x 64) into one TRANSPOSED 128 x K fp16
// weight: dst[j][k] = (j < 64) ? w41[k][j] : w42[k][j-64].
__global__ __launch_bounds__(256) void pack_w4142_t(const float* __restrict__ w41,
                                                    const float* __restrict__ w42,
                                                    _Float16* __restrict__ dst, int K) {
  int i = blockIdx.x * blockDim.x + threadIdx.x;
  if (i >= 128 * K) return;
  int j = i / K, k = i - j * K;
  float v = (j < 64) ? w41[k * 64 + j] : w42[k * 64 + (j - 64)];
  dst[i] = (_Float16)v;
}

// Plain fp32 -> fp16 converter for the batch input x.
__global__ __launch_bounds__(256) void f32_to_f16_v4(const float* __restrict__ src,
                                                     _Float16* __restrict__ dst, int n4) {
  int i = blockIdx.x * blockDim.x + threadIdx.x;
  if (i >= n4) return;
  v4f s = *(const v4f*)(src + 4 * (size_t)i);
  v4h h;
  h[0] = (_Float16)s[0]; h[1] = (_Float16)s[1];
  h[2] = (_Float16)s[2]; h[3] = (_Float16)s[3];
  *(v4h*)(dst + 4 * (size_t)i) = h;
}

// ---------------------------------------------------------------------------
// WMMA GEMM: C[M,N] = A[M,K] (f16 row-major) * Wt[N,K] (f16, pre-transposed)
// Block: 256 threads (8 waves). WG tile 256x128, wave tile 64x64 (4x4 WMMAs
// -> 16 WMMAs per 16 ds_load_b128 fragment reads per K-step). K stepped by 32
// with double-buffered LDS fed by global_load_async_to_lds_b128.
// All layer shapes satisfy M % 256 == 0, N % 128 == 0, K % 32 == 0 -> no
// guards, EXEC stays all-ones (required by WMMA).
//
// ACT == 0: out = relu(C + b0) as fp16 -> outH. Epilogue staged through a
//           wave-private LDS buffer so global stores are full-line b128s.
// ACT == 1: out = eps + softplus(C + bias), column-split at `splitN`:
//           cols [0,splitN) -> outF0 (bias b0), cols [splitN,N) -> outF1 (bias b1)
// ---------------------------------------------------------------------------
#define LDT 40  // padded LDS row stride: 32 data + 8 pad halves (bank-conflict-free)

template <int ACT>
__global__ __launch_bounds__(256)
void gemm_wmma_kernel(const _Float16* __restrict__ A, const _Float16* __restrict__ Wt,
                      const float* __restrict__ b0, const float* __restrict__ b1,
                      _Float16* __restrict__ outH, float* __restrict__ outF0,
                      float* __restrict__ outF1, int M, int N, int K, int splitN) {
  __shared__ _Float16 lA[2][256 * LDT];   // A tile: 256 rows x 32 K-halves
  __shared__ _Float16 lB[2][128 * LDT];   // B tile: 128 cols x 32 K-halves

  const int tid  = threadIdx.x;
  const int wave = tid >> 5;
  const int lane = tid & 31;
  const int wm   = wave & 3;   // 4 waves along M (4 * 64 = 256)
  const int wn   = wave >> 2;  // 2 waves along N (2 * 64 = 128)
  const int m0   = blockIdx.y * 256;
  const int n0   = blockIdx.x * 128;
  const int l15  = lane & 15;
  const int lhi  = lane >> 4;

  v8f acc[4][4];
#pragma unroll
  for (int mi = 0; mi < 4; ++mi)
#pragma unroll
    for (int nj = 0; nj < 4; ++nj) acc[mi][nj] = v8f{};

  // Per K-step each thread moves 4 A-chunks + 2 B-chunks of 8 halves (b128).
  const int rA = tid >> 2;               // 0..63 (+64/128/192 for A, +64 for B)
  const int cA = (tid & 3) * 8;          // 0,8,16,24
  auto loadTiles = [&](int kk, int buf) {
#if USE_ASYNC_LDS
#pragma unroll
    for (int i = 0; i < 4; ++i) {
      int r = rA + i * 64;
      __builtin_amdgcn_global_load_async_to_lds_b128(
          (gptr_b128)(A + (size_t)(m0 + r) * K + kk + cA),
          (lptr_b128)&lA[buf][r * LDT + cA], 0, 0);
    }
#pragma unroll
    for (int i = 0; i < 2; ++i) {
      int r = rA + i * 64;
      __builtin_amdgcn_global_load_async_to_lds_b128(
          (gptr_b128)(Wt + (size_t)(n0 + r) * K + kk + cA),
          (lptr_b128)&lB[buf][r * LDT + cA], 0, 0);
    }
#else
    v8h va[4], vb[2];
#pragma unroll
    for (int i = 0; i < 4; ++i)
      va[i] = *(const v8h*)(A + (size_t)(m0 + rA + i * 64) * K + kk + cA);
#pragma unroll
    for (int i = 0; i < 2; ++i)
      vb[i] = *(const v8h*)(Wt + (size_t)(n0 + rA + i * 64) * K + kk + cA);
#pragma unroll
    for (int i = 0; i < 4; ++i)
      *(v8h*)&lA[buf][(rA + i * 64) * LDT + cA] = va[i];
#pragma unroll
    for (int i = 0; i < 2; ++i)
      *(v8h*)&lB[buf][(rA + i * 64) * LDT + cA] = vb[i];
#endif
  };

  auto compute = [&](int buf) {
    // A fragment: lane holds row (lane&15); K = [kh..kh+7] ++ [16+kh..16+kh+7]
    v16h af[4];
#pragma unroll
    for (int mi = 0; mi < 4; ++mi) {
      int r  = wm * 64 + mi * 16 + l15;
      int kh = lhi * 8;
      v8h lo = *(const v8h*)&lA[buf][r * LDT + kh];
      v8h hi = *(const v8h*)&lA[buf][r * LDT + 16 + kh];
#pragma unroll
      for (int e = 0; e < 8; ++e) { af[mi][e] = lo[e]; af[mi][8 + e] = hi[e]; }
    }
    // B fragment: lane holds col (lane&15); K = [kb..kb+7] ++ [16+kb..16+kb+7] (weights pre-transposed)
    v16h bf[4];
#pragma unroll
    for (int nj = 0; nj < 4; ++nj) {
      int c  = wn * 64 + nj * 16 + l15;
      int kb = lhi * 8;
      v8h lo = *(const v8h*)&lB[buf][c * LDT + kb];
      v8h hi = *(const v8h*)&lB[buf][c * LDT + 16 + kb];
#pragma unroll
      for (int e = 0; e < 8; ++e) { bf[nj][e] = lo[e]; bf[nj][8 + e] = hi[e]; }
    }
#pragma unroll
    for (int mi = 0; mi < 4; ++mi)
#pragma unroll
      for (int nj = 0; nj < 4; ++nj)
        acc[mi][nj] = __builtin_amdgcn_wmma_f32_16x16x32_f16(
            false, af[mi], false, bf[nj], (short)0, acc[mi][nj], false, false);
  };

  const int nk = K / 32;
  loadTiles(0, 0);
  for (int t = 0; t < nk; ++t) {
    int buf = t & 1;
    if (t + 1 < nk) {
      loadTiles((t + 1) * 32, buf ^ 1);  // overwrites buffer drained at t-1
#if USE_ASYNC_LDS
      __builtin_amdgcn_s_wait_asynccnt(6);  // tile t landed; tile t+1 may fly
#endif
    } else {
#if USE_ASYNC_LDS
      __builtin_amdgcn_s_wait_asynccnt(0);
#endif
    }
    __syncthreads();      // tile t visible to all waves
    compute(buf);
    __syncthreads();      // everyone done reading buf before it is overwritten
  }
  // NOTE: the trailing __syncthreads() above also means lA/lB are free to be
  // reused as epilogue staging space.

  // ---- Epilogue. C/D layout: VGPR g -> row g (lanes 0-15) / row 8+g (16-31)
  if (ACT == 0) {
    // Bias is mi-invariant: hoist.
    float bias[4];
#pragma unroll
    for (int nj = 0; nj < 4; ++nj) bias[nj] = b0[n0 + wn * 64 + nj * 16 + l15];

    // Wave-private 16x64 fp16 staging tile in (drained) lB: scatter strided
    // b16 writes into LDS, then emit fully-coalesced b128 global stores
    // (each store instruction covers 4 complete 128B cache lines).
    _Float16* stage = ((_Float16*)lB) + wave * (16 * 64);
#pragma unroll
    for (int mi = 0; mi < 4; ++mi) {
#pragma unroll
      for (int nj = 0; nj < 4; ++nj) {
        int cs = nj * 16 + l15;
        int rs = lhi ? 8 : 0;
#pragma unroll
        for (int g = 0; g < 8; ++g) {
          float v = acc[mi][nj][g] + bias[nj];
          stage[(rs + g) * 64 + cs] = (_Float16)fmaxf(v, 0.f);
        }
      }
      // Same-wave LDS RAW: DS ops from one wave complete in order; the
      // compiler inserts the required s_wait_dscnt. No cross-wave sharing.
      const int mrow0 = m0 + wm * 64 + mi * 16;
      const int ncol0 = n0 + wn * 64;
#pragma unroll
      for (int j = 0; j < 4; ++j) {
        int q  = j * 32 + lane;         // 128 chunks of 8 halves (16 rows x 8)
        int r  = q >> 3;
        int co = (q & 7) * 8;
        v8h vv = *(const v8h*)&stage[r * 64 + co];
        *(v8h*)(outH + (size_t)(mrow0 + r) * N + ncol0 + co) = vv;
      }
    }
  } else {
#pragma unroll
    for (int mi = 0; mi < 4; ++mi) {
#pragma unroll
      for (int nj = 0; nj < 4; ++nj) {
        int colt = n0 + wn * 64 + nj * 16 + l15;
        int rbase = m0 + wm * 64 + mi * 16 + (lhi ? 8 : 0);
#pragma unroll
        for (int g = 0; g < 8; ++g) {
          int row = rbase + g;
          float v = acc[mi][nj][g];
          float bb = (colt < splitN) ? b0[colt] : b1[colt - splitN];
          v += bb;
          float sp = (v > 20.f) ? v : log1pf(__expf(v));
          float o = EPSF + sp;
          if (colt < splitN)
            outF0[(size_t)row * splitN + colt] = o;
          else
            outF1[(size_t)row * (N - splitN) + (colt - splitN)] = o;
        }
      }
    }
  }
}

// ---------------------------------------------------------------------------
// Marsaglia-Tsang gamma rejection sampler (alpha-boosted), 24 fixed rounds.
// ---------------------------------------------------------------------------
__device__ __forceinline__ uint32_t pcg_hash(uint32_t s) {
  s = s * 747796405u + 2891336453u;
  uint32_t w = ((s >> ((s >> 28) + 4u)) ^ s) * 277803737u;
  return (w >> 22) ^ w;
}

__global__ __launch_bounds__(256) void gamma_sampler(const float* __restrict__ la,
                                                     const float* __restrict__ lb,
                                                     float* __restrict__ z,
                                                     _Float16* __restrict__ zh, int n) {
  int i = blockIdx.x * blockDim.x + threadIdx.x;
  if (i >= n) return;
  float al = la[i], be = lb[i];
  float d = (al + 1.f) - (1.f / 3.f);
  float c = __frsqrt_rn(9.f * d);

  float eps_s = 0.f, u_s = 0.5f;
  bool done = false;
  uint32_t st = 0x9E3779B9u ^ (uint32_t)i;
#pragma unroll 4
  for (int r = 0; r < 24; ++r) {
    uint32_t h1 = pcg_hash(st + (uint32_t)r * 0x85EBCA6Bu);
    uint32_t h2 = pcg_hash(h1 ^ 0xC2B2AE35u);
    uint32_t h3 = pcg_hash(h2 + 0x27D4EB2Fu);
    float u1 = (float)(h1 >> 8) * (1.f / 16777216.f) + 1e-7f;
    float u2 = (float)(h2 >> 8) * (1.f / 16777216.f);
    float u  = (float)(h3 >> 8) * (1.f / 16777216.f) + 1e-7f;
    float eps = __fsqrt_rn(-2.f * __logf(u1)) * __cosf(6.2831853f * u2);
    float v  = 1.f + c * eps;
    float v3 = v * v * v;
    float sv3 = (v > 0.f) ? v3 : 1.f;
    float e2 = eps * eps;
    bool acc = (v > 0.f) &&
               ((u < 1.f - 0.0331f * e2 * e2) ||
                (__logf(u) < 0.5f * e2 + d * (1.f - sv3 + __logf(sv3))));
    if (!done && acc) { eps_s = eps; u_s = u; done = true; }
  }
  float vs = 1.f + c * eps_s;
  vs = vs * vs * vs;
  float zz = __expf(__logf(d * vs + EPSF) + __logf(u_s + EPSF) / (al + EPSF)) / (be + EPSF);
  z[i] = zz;
  zh[i] = (_Float16)zz;
}

// ---------------------------------------------------------------------------
// Host orchestration
// ---------------------------------------------------------------------------
extern "C" void kernel_launch(void* const* d_in, const int* in_sizes, int n_in,
                              void* d_out, int out_size, void* d_ws, size_t ws_size,
                              hipStream_t stream) {
  (void)in_sizes; (void)n_in; (void)out_size; (void)ws_size;

  const int B = 8192, D = 2048, L = 64;
  const int E0 = 1024, E1 = 1024, E2 = 2048;

  const float* x      = (const float*)d_in[0];
  const float* fc1_w  = (const float*)d_in[1];  const float* fc1_b  = (const float*)d_in[2];
  const float* fc2_w  = (const float*)d_in[3];  const float* fc2_b  = (const float*)d_in[4];
  const float* fc3_w  = (const float*)d_in[5];  const float* fc3_b  = (const float*)d_in[6];
  const float* fc41_w = (const float*)d_in[7];  const float* fc41_b = (const float*)d_in[8];
  const float* fc42_w = (const float*)d_in[9];  const float* fc42_b = (const float*)d_in[10];
  const float* fc4_w  = (const float*)d_in[11]; const float* fc4_b  = (const float*)d_in[12];
  const float* fc5_w  = (const float*)d_in[13]; const float* fc5_b  = (const float*)d_in[14];
  const float* fc6_w  = (const float*)d_in[15]; const float* fc6_b  = (const float*)d_in[16];
  const float* fc7_w  = (const float*)d_in[17]; const float* fc7_b  = (const float*)d_in[18];

  float* out = (float*)d_out;
  const size_t RA_OFF = 0;
  const size_t RB_OFF = (size_t)B * D;                 // 16,777,216
  const size_t LA_OFF = RB_OFF + (size_t)B * D;        // 33,554,432
  const size_t LB_OFF = LA_OFF + (size_t)B * L;        // 34,078,720
  const size_t Z_OFF  = LB_OFF + (size_t)B * L;        // 34,603,008

  // workspace carve-out (fp16 buffers), 256B aligned
  size_t off = 0;
  auto take = [&](size_t elems) -> _Float16* {
    _Float16* p = (_Float16*)((char*)d_ws + off);
    off += ((elems * 2 + 255) / 256) * 256;
    return p;
  };
  _Float16* bufA = take((size_t)B * D);      // x_f16, then h2, h4, h6
  _Float16* bufB = take((size_t)B * D);      // h1, h3, h5
  _Float16* zh   = take((size_t)B * L);
  _Float16* w1t  = take((size_t)E0 * D);     // all weights stored TRANSPOSED [N][K]
  _Float16* w2t  = take((size_t)E1 * E0);
  _Float16* w3t  = take((size_t)E2 * E1);
  _Float16* w41t = take((size_t)128 * E2);
  _Float16* w4t  = take((size_t)E2 * L);
  _Float16* w5t  = take((size_t)E1 * E2);
  _Float16* w6t  = take((size_t)E0 * E1);
  _Float16* w7t  = take((size_t)(2 * D) * E0);

  // fp32 -> fp16 with transpose for weights; straight convert for x
  auto convT = [&](const float* s, _Float16* dst, int K, int N) {
    transpose_f32_to_f16<<<dim3(K / 32, N / 32), 256, 0, stream>>>(s, dst, K, N);
  };
  {
    int n4 = (B * D) / 4;
    f32_to_f16_v4<<<(n4 + 255) / 256, 256, 0, stream>>>(x, bufA, n4);
  }
  convT(fc1_w, w1t, D, E0);
  convT(fc2_w, w2t, E0, E1);
  convT(fc3_w, w3t, E1, E2);
  convT(fc4_w, w4t, L, E2);
  convT(fc5_w, w5t, E2, E1);
  convT(fc6_w, w6t, E1, E0);
  convT(fc7_w, w7t, E0, 2 * D);
  pack_w4142_t<<<(128 * E2 + 255) / 256, 256, 0, stream>>>(fc41_w, fc42_w, w41t, E2);

  auto gemm_relu = [&](const _Float16* Ai, const _Float16* Wi, const float* bi,
                       _Float16* Oi, int M, int N, int K) {
    dim3 grid(N / 128, M / 256);
    gemm_wmma_kernel<0><<<grid, 256, 0, stream>>>(Ai, Wi, bi, bi, Oi, nullptr, nullptr,
                                                  M, N, K, N);
  };
  auto gemm_split_sp = [&](const _Float16* Ai, const _Float16* Wi, const float* bi0,
                           const float* bi1, float* O0, float* O1, int M, int N, int K,
                           int splitN) {
    dim3 grid(N / 128, M / 256);
    gemm_wmma_kernel<1><<<grid, 256, 0, stream>>>(Ai, Wi, bi0, bi1, nullptr, O0, O1,
                                                  M, N, K, splitN);
  };

  // encoder
  gemm_relu(bufA, w1t, fc1_b, bufB, B, E0, D);    // h1 = relu(x @ W1 + b1)
  gemm_relu(bufB, w2t, fc2_b, bufA, B, E1, E0);   // h2  (overwrites x copy)
  gemm_relu(bufA, w3t, fc3_b, bufB, B, E2, E1);   // h3
  // logalpha / logbeta (fused N=128, split at 64)
  gemm_split_sp(bufB, w41t, fc41_b, fc42_b, out + LA_OFF, out + LB_OFF,
                B, 128, E2, 64);
  // gamma reparameterization
  gamma_sampler<<<((B * L) + 255) / 256, 256, 0, stream>>>(out + LA_OFF, out + LB_OFF,
                                                           out + Z_OFF, zh, B * L);
  // decoder
  gemm_relu(zh,   w4t, fc4_b, bufA, B, E2, L);    // h4
  gemm_relu(bufA, w5t, fc5_b, bufB, B, E1, E2);   // h5
  gemm_relu(bufB, w6t, fc6_b, bufA, B, E0, E1);   // h6
  // dec = h6 @ W7 + b7 -> softplus split into recon_al | recon_be
  gemm_split_sp(bufA, w7t, fc7_b, fc7_b + D, out + RA_OFF, out + RB_OFF,
                B, 2 * D, E0, D);
}
